// DynamicSincConv1d_90821378441579
// MI455X (gfx1250) — matrix operand order repaired and checked by reading, not compiled, not run on hardware
//
#include <hip/hip_runtime.h>
#include <hip/hip_bf16.h>

typedef __attribute__((ext_vector_type(16))) _Float16 v16h;
typedef __attribute__((ext_vector_type(8)))  float    v8f;

#define PI_F      3.14159265358979323846f
#define C2PI_256  0.0245436926f   /* 2*pi/256 */
#define C4PI_256  0.0490873852f   /* 4*pi/256 */

#define BDIM   8
#define ICH    2
#define OCH    4
#define NS     4
#define CCH    32
#define KWIN   256
#define HOP    64
#define TLEN   65536
#define NF     1025
#define FPAD   1056          /* 33 * 32 */
#define NDFT   272           /* 258 interleaved re/im cols padded to 17*16 */
#define NGPAD  288           /* irfft reduction 258 padded to 9*32 */
#define SPAN   4288          /* 63*64 + 256 samples per channel per 64-frame tile */

/* ---- workspace layout (bytes, 256-aligned) ---- */
#define OFF_W1H  ((size_t)0)          /* f16 [32][8192]        524288  */
#define OFF_DFT  ((size_t)524288)     /* f16 [272][256]        139264  */
#define OFF_GT   ((size_t)663552)     /* f16 [256][288]        147456  */
#define OFF_H    ((size_t)811008)     /* f16 [8][32][1056]     540672  */
#define OFF_AMP  ((size_t)1351680)    /* f32 [256][1025]      1049600  */
#define OFF_WID  ((size_t)2401280)    /* f32 [256][1025]      1049600  */
#define OFF_FF   ((size_t)3450880)    /* f16 [64*1056][272]  36765696  */
#define OFF_X    ((size_t)40216576)   /* f16 [16*1056][272]   9191424  */
#define OFF_FR   ((size_t)49408000)   /* f16 [32*1056][256]  17301504  */

__device__ __forceinline__ float sinc1(float xv) {
  float px = PI_F * xv;
  if (fabsf(px) < 1e-4f) return 1.0f - px * px * (1.0f / 6.0f);
  return __sinf(px) / px;
}

__device__ __forceinline__ float ola_w(int k) {
  return 0.5f * (1.0f - __cosf((float)k * C2PI_256));
}

/* ================= prep kernels ================= */
__global__ void prep_w1(const float* __restrict__ w1, _Float16* __restrict__ w1h) {
  int i = blockIdx.x * 256 + threadIdx.x;
  if (i < 32 * 8192) w1h[i] = (_Float16)w1[i];
}

/* DFT^T: row n (interleaved re/im spectral col), col k.  n=2f: cos, n=2f+1: -sin */
__global__ void prep_dft(_Float16* __restrict__ dftt) {
  int i = blockIdx.x * 256 + threadIdx.x;
  if (i >= NDFT * KWIN) return;
  int n = i / KWIN, k = i % KWIN;
  float v = 0.0f;
  if (n < 258) {
    int fb = n >> 1;
    float th = (float)((k * fb) & 255) * C2PI_256;
    v = (n & 1) ? -__sinf(th) : __cosf(th);
  }
  dftt[i] = (_Float16)v;
}

/* iDFT^T: row k (output sample), col n (interleaved re/im), with rfft weights */
__global__ void prep_gt(_Float16* __restrict__ gt) {
  int i = blockIdx.x * 256 + threadIdx.x;
  if (i >= KWIN * NGPAD) return;
  int k = i / NGPAD, n = i % NGPAD;
  float v = 0.0f;
  if (n < 258) {
    int fb = n >> 1;
    float w = (fb == 0 || fb == 128) ? (1.0f / 256.0f) : (2.0f / 256.0f);
    float th = (float)((k * fb) & 255) * C2PI_256;
    v = (n & 1) ? -w * __sinf(th) : w * __cosf(th);
  }
  gt[i] = (_Float16)v;
}

/* ===== stage 1: strided conv as WMMA GEMM, channel span staged in LDS ===== */
__global__ __launch_bounds__(256) void k_condgemm(
    const float* __restrict__ cond, const _Float16* __restrict__ w1h,
    const float* __restrict__ b1, _Float16* __restrict__ hbuf) {
  __shared__ __align__(16) _Float16 sc[SPAN + 64];
  int b = blockIdx.y;
  int n0 = blockIdx.x * 64;
  int base = n0 * HOP - 128;          /* first sample of the tile span */
  int wave = threadIdx.x >> 5, lane = threadIdx.x & 31;
  int mt = wave >> 2, nt = wave & 3;
  int lh = (lane & 16) ? 1 : 0;
  int l15 = lane & 15;
  int fcol = n0 + nt * 16 + l15;
  const float* src_b = cond + (long)b * CCH * TLEN;
  const _Float16* arow_p = w1h + (mt * 16 + l15) * 8192;
  v8f acc = {};
  for (int ci = 0; ci < CCH; ++ci) {
    __syncthreads();                  /* protect previous iteration's reads */
    const float* sp = src_b + ci * TLEN;
    /* cooperative zero-padded load of the 4288-sample span, f32 -> f16 */
    for (int cq = threadIdx.x; cq < SPAN / 4; cq += 256) {
      int s0 = base + cq * 4;
      float4 q;
      if (s0 >= 0 && s0 <= TLEN - 4) {
        q = *(const float4*)(sp + s0);
        __builtin_prefetch(sp + TLEN + s0, 0, 1);   /* next channel's span */
      } else {
        float* qa = (float*)&q;
        #pragma unroll
        for (int j = 0; j < 4; ++j) {
          int sj = s0 + j;
          qa[j] = (sj >= 0 && sj < TLEN) ? sp[sj] : 0.0f;
        }
      }
      _Float16 h4[4];
      h4[0] = (_Float16)q.x; h4[1] = (_Float16)q.y;
      h4[2] = (_Float16)q.z; h4[3] = (_Float16)q.w;
      *(uint2*)&sc[cq * 4] = *(const uint2*)h4;
    }
    __syncthreads();
    /* 8 branch-free WMMA chunks over this channel's K range */
    #pragma unroll
    for (int cc = 0; cc < 8; ++cc) {
      v16h av, bv;
      int kb = (ci * 8 + cc) * 32 + lh * 16;
      ((uint4*)&av)[0] = *(const uint4*)(arow_p + kb);
      ((uint4*)&av)[1] = *(const uint4*)(arow_p + kb + 8);
      int k = cc * 32 + lh * 16;
      const _Float16* bp = &sc[(nt * 16 + l15) * HOP + k];
      ((uint4*)&bv)[0] = *(const uint4*)bp;
      ((uint4*)&bv)[1] = *(const uint4*)(bp + 8);
      acc = __builtin_amdgcn_wmma_f32_16x16x32_f16(false, av, false, bv,
                                                   (short)0, acc, false, false);
    }
  }
  if (fcol <= 1024) {
    #pragma unroll
    for (int r = 0; r < 8; ++r) {
      int m = mt * 16 + r + lh * 8;
      float v = acc[r] + b1[m];
      v = v >= 0.0f ? v : 0.01f * v;
      hbuf[(b * CCH + m) * FPAD + fcol] = (_Float16)v;
    }
  }
}

/* ============ stage 2: 1x1 conv + tanh -> amp / width ============ */
__global__ __launch_bounds__(256) void k_params(
    const _Float16* __restrict__ hbuf, const float* __restrict__ w2,
    const float* __restrict__ b2, float* __restrict__ ampb, float* __restrict__ widb) {
  int gid = blockIdx.x * 256 + threadIdx.x;
  if (gid >= BDIM * NF * 64) return;
  int c2 = gid % 64;
  int rest = gid / 64;
  int f = rest % NF;
  int b = rest / NF;
  float acc = b2[c2];
  const _Float16* hp = hbuf + (b * CCH) * FPAD + f;
  #pragma unroll 8
  for (int ci = 0; ci < CCH; ++ci)
    acc += w2[c2 * CCH + ci] * (float)hp[ci * FPAD];
  float v = tanhf(acc);
  int s = c2 & 3, i = (c2 >> 2) & 1, o = (c2 >> 3) & 3, p = c2 >> 5;
  int bois = ((b * OCH + o) * ICH + i) * NS + s;
  (p == 0 ? ampb : widb)[bois * NF + f] = v;
}

/* ============ stage 3: sinc FIR synthesis (LDS) + rfft via WMMA ============ */
__global__ __launch_bounds__(256) void k_sincfft(
    const float* __restrict__ ampb, const float* __restrict__ widb,
    const _Float16* __restrict__ dftt, _Float16* __restrict__ ff) {
  __shared__ __align__(16) _Float16 sf[32 * 264];
  int g = blockIdx.y;          /* (b*O+o)*I+i : 0..63 */
  int tile = blockIdx.x;       /* 0..32       */
  int tid = threadIdx.x;
  {
    int row = tid >> 3, kg = tid & 7;
    int f = tile * 32 + row;
    bool vf = f < NF;
    float a[4], w[4];
    #pragma unroll
    for (int s = 0; s < 4; ++s) {
      int idx = (g * NS + s) * NF + f;
      a[s] = vf ? ampb[idx] : 0.0f;
      w[s] = vf ? widb[idx] : 0.0f;
    }
    for (int kk = 0; kk < 32; ++kk) {
      int k = kg * 32 + kk;
      float t = (float)(k - 128) * (1.0f / 256.0f);
      float av = 0.0f;
      #pragma unroll
      for (int s = 0; s < 4; ++s) av += a[s] * sinc1(w[s] * t + 1e-6f);
      float fir = 0.42f - 0.5f * __cosf((float)k * C2PI_256)
                        + 0.08f * __cosf((float)k * C4PI_256);
      sf[row * 264 + k] = (_Float16)(av * fir * (1.0f / 1024.0f));
    }
  }
  __syncthreads();
  int wave = tid >> 5, lane = tid & 31;
  int mt = wave >> 2, nt = wave & 3;
  int lh = (lane & 16) ? 1 : 0, l15 = lane & 15;
  for (int nb = 0; nb < NDFT; nb += 64) {
    int ncb = nb + nt * 16;
    if (ncb >= NDFT) continue;          /* wave-uniform */
    int ncol = ncb + l15;
    v8f acc = {};
    #pragma unroll
    for (int c = 0; c < 8; ++c) {
      v16h av, bv;
      const _Float16* ap = &sf[(mt * 16 + l15) * 264 + c * 32 + lh * 16];
      ((uint4*)&av)[0] = *(const uint4*)ap;
      ((uint4*)&av)[1] = *(const uint4*)(ap + 8);
      const _Float16* bp = dftt + ncol * KWIN + c * 32 + lh * 16;
      ((uint4*)&bv)[0] = *(const uint4*)bp;
      ((uint4*)&bv)[1] = *(const uint4*)(bp + 8);
      acc = __builtin_amdgcn_wmma_f32_16x16x32_f16(false, av, false, bv,
                                                   (short)0, acc, false, false);
    }
    #pragma unroll
    for (int r = 0; r < 8; ++r) {
      int m = mt * 16 + r + lh * 8;
      long rowg = (long)g * FPAD + tile * 32 + m;
      ff[rowg * NDFT + ncol] = (_Float16)acc[r];
    }
  }
}

/* ============ stage 4: STFT (windowed frames in LDS) via WMMA ============ */
__global__ __launch_bounds__(256) void k_stft(
    const float* __restrict__ x, const _Float16* __restrict__ dftt,
    _Float16* __restrict__ X) {
  __shared__ __align__(16) _Float16 sx[32 * 264];
  int g = blockIdx.y;          /* b*I+i : 0..15 */
  int tile = blockIdx.x;
  int tid = threadIdx.x;
  {
    int row = tid >> 3, kg = tid & 7;
    int f = tile * 32 + row;
    const float* xp = x + (long)g * TLEN;
    for (int kk = 0; kk < 32; ++kk) {
      int k = kg * 32 + kk;
      float v = 0.0f;
      if (f < NF) {
        int j = f * HOP + k - 128;
        if (j < 0) j = -j;
        if (j > TLEN - 1) j = 2 * (TLEN - 1) - j;   /* reflect */
        v = xp[j] * ola_w(k);
      }
      sx[row * 264 + k] = (_Float16)v;
    }
  }
  __syncthreads();
  int wave = tid >> 5, lane = tid & 31;
  int mt = wave >> 2, nt = wave & 3;
  int lh = (lane & 16) ? 1 : 0, l15 = lane & 15;
  for (int nb = 0; nb < NDFT; nb += 64) {
    int ncb = nb + nt * 16;
    if (ncb >= NDFT) continue;
    int ncol = ncb + l15;
    v8f acc = {};
    #pragma unroll
    for (int c = 0; c < 8; ++c) {
      v16h av, bv;
      const _Float16* ap = &sx[(mt * 16 + l15) * 264 + c * 32 + lh * 16];
      ((uint4*)&av)[0] = *(const uint4*)ap;
      ((uint4*)&av)[1] = *(const uint4*)(ap + 8);
      const _Float16* bp = dftt + ncol * KWIN + c * 32 + lh * 16;
      ((uint4*)&bv)[0] = *(const uint4*)bp;
      ((uint4*)&bv)[1] = *(const uint4*)(bp + 8);
      acc = __builtin_amdgcn_wmma_f32_16x16x32_f16(false, av, false, bv,
                                                   (short)0, acc, false, false);
    }
    #pragma unroll
    for (int r = 0; r < 8; ++r) {
      int m = mt * 16 + r + lh * 8;
      long rowg = (long)g * FPAD + tile * 32 + m;
      X[rowg * NDFT + ncol] = (_Float16)acc[r];
    }
  }
}

/* ==== stage 5+6: complex mult + channel reduce (LDS) + irfft*ola via WMMA ==== */
__global__ __launch_bounds__(256) void k_isynth(
    const _Float16* __restrict__ X, const _Float16* __restrict__ ff,
    const _Float16* __restrict__ gt, _Float16* __restrict__ fr) {
  __shared__ __align__(16) _Float16 sy[32 * 296];
  int g = blockIdx.y;          /* b*O+o : 0..31 */
  int b = g >> 2;
  int tile = blockIdx.x;
  int tid = threadIdx.x;
  {
    int row = tid >> 3, pg = tid & 7;
    int f = tile * 32 + row;
    for (int p = 0; p < 17; ++p) {
      int pair = pg * 17 + p;            /* 0..135 */
      float yr = 0.0f, yi = 0.0f;
      if (f < NF && pair < 129) {
        #pragma unroll
        for (int i = 0; i < ICH; ++i) {
          long xo = ((long)(b * ICH + i) * FPAD + f) * NDFT + 2 * pair;
          long fo = ((long)(g * ICH + i) * FPAD + f) * NDFT + 2 * pair;
          float xr = (float)X[xo], xi = (float)X[xo + 1];
          float hr = (float)ff[fo], hi = (float)ff[fo + 1];
          yr += xr * hr - xi * hi;
          yi += xr * hi + xi * hr;
        }
      }
      sy[row * 296 + 2 * pair]     = (_Float16)yr;
      sy[row * 296 + 2 * pair + 1] = (_Float16)yi;
    }
    for (int n = 272 + pg; n < NGPAD; n += 8) sy[row * 296 + n] = (_Float16)0.0f;
  }
  __syncthreads();
  int wave = tid >> 5, lane = tid & 31;
  int mt = wave >> 2, nt = wave & 3;
  int lh = (lane & 16) ? 1 : 0, l15 = lane & 15;
  for (int nb = 0; nb < 256; nb += 64) {
    int ncol = nb + nt * 16 + l15;       /* output sample k: 0..255 */
    v8f acc = {};
    #pragma unroll
    for (int c = 0; c < 9; ++c) {
      v16h av, bv;
      const _Float16* ap = &sy[(mt * 16 + l15) * 296 + c * 32 + lh * 16];
      ((uint4*)&av)[0] = *(const uint4*)ap;
      ((uint4*)&av)[1] = *(const uint4*)(ap + 8);
      const _Float16* bp = gt + ncol * NGPAD + c * 32 + lh * 16;
      ((uint4*)&bv)[0] = *(const uint4*)bp;
      ((uint4*)&bv)[1] = *(const uint4*)(bp + 8);
      acc = __builtin_amdgcn_wmma_f32_16x16x32_f16(false, av, false, bv,
                                                   (short)0, acc, false, false);
    }
    float w = ola_w(ncol);
    #pragma unroll
    for (int r = 0; r < 8; ++r) {
      int m = mt * 16 + r + lh * 8;
      long rowg = (long)g * FPAD + tile * 32 + m;
      fr[rowg * KWIN + ncol] = (_Float16)(acc[r] * w);
    }
  }
}

/* ============ stage 7: gather overlap-add + env normalize + bias ============ */
__global__ __launch_bounds__(256) void k_ola(
    const _Float16* __restrict__ fr, const float* __restrict__ bias,
    float* __restrict__ out) {
  int gid = blockIdx.x * 256 + threadIdx.x;
  if (gid >= BDIM * OCH * TLEN) return;
  int t = gid % TLEN;
  int bo = gid / TLEN;
  int o = bo % OCH;
  int s = t + 128;
  int fh = s >> 6; if (fh > NF - 1) fh = NF - 1;
  int fl = (s >= 255) ? ((s - 255 + 63) >> 6) : 0;
  float y = 0.0f, e = 0.0f;
  for (int f = fl; f <= fh; ++f) {
    int k = s - (f << 6);
    float w = ola_w(k);
    y += (float)fr[((long)bo * FPAD + f) * KWIN + k];
    e += w * w;
  }
  out[gid] = y / e + bias[o];
}

extern "C" void kernel_launch(void* const* d_in, const int* in_sizes, int n_in,
                              void* d_out, int out_size, void* d_ws, size_t ws_size,
                              hipStream_t stream) {
  const float* x    = (const float*)d_in[0];
  const float* cond = (const float*)d_in[1];
  const float* w1   = (const float*)d_in[2];
  const float* b1   = (const float*)d_in[3];
  const float* w2   = (const float*)d_in[4];
  const float* b2   = (const float*)d_in[5];
  const float* bias = (const float*)d_in[6];
  float* out = (float*)d_out;
  char* ws = (char*)d_ws;

  _Float16* w1h  = (_Float16*)(ws + OFF_W1H);
  _Float16* dftt = (_Float16*)(ws + OFF_DFT);
  _Float16* gt   = (_Float16*)(ws + OFF_GT);
  _Float16* hbuf = (_Float16*)(ws + OFF_H);
  float*    ampb = (float*)(ws + OFF_AMP);
  float*    widb = (float*)(ws + OFF_WID);
  _Float16* ff   = (_Float16*)(ws + OFF_FF);
  _Float16* Xb   = (_Float16*)(ws + OFF_X);
  _Float16* frb  = (_Float16*)(ws + OFF_FR);

  prep_w1 <<<1024, 256, 0, stream>>>(w1, w1h);
  prep_dft<<<(NDFT * KWIN + 255) / 256, 256, 0, stream>>>(dftt);
  prep_gt <<<(KWIN * NGPAD + 255) / 256, 256, 0, stream>>>(gt);

  k_condgemm<<<dim3(17, BDIM), 256, 0, stream>>>(cond, w1h, b1, hbuf);
  k_params  <<<(BDIM * NF * 64 + 255) / 256, 256, 0, stream>>>(hbuf, w2, b2, ampb, widb);
  k_sincfft <<<dim3(33, BDIM * OCH * ICH), 256, 0, stream>>>(ampb, widb, dftt, ff);
  k_stft    <<<dim3(33, BDIM * ICH), 256, 0, stream>>>(x, dftt, Xb);
  k_isynth  <<<dim3(33, BDIM * OCH), 256, 0, stream>>>(Xb, ff, gt, frb);
  k_ola     <<<(BDIM * OCH * TLEN + 255) / 256, 256, 0, stream>>>(frb, bias, out);
}
